// ResnetPointHeightnet_63788854280412
// MI455X (gfx1250) — compile-verified, matrix operand order checked by reading, not run on hardware
//
#include <hip/hip_runtime.h>

// ---------------------------------------------------------------------------
// Types for CDNA5 WMMA (gfx1250, wave32).
// ---------------------------------------------------------------------------
typedef __bf16 bf16_t;
typedef __bf16 v16bf __attribute__((ext_vector_type(16)));
typedef __bf16 v8bf  __attribute__((ext_vector_type(8)));
typedef float  v8f   __attribute__((ext_vector_type(8)));

#define B_   4
#define T_   32768
#define M_   (B_ * T_)        // 131072 points
#define HID_ 128
#define GRID_ 64
#define NCH_ 4
#define LPL_ (NCH_ + 1)       // 5 planes (incl. zero plane)
#define CDIM_ 128
#define COUT_ELEMS ((size_t)B_ * LPL_ * GRID_ * GRID_ * CDIM_)  // 10485760

// Order-preserving float<->uint encoding for atomic max (handles negatives).
__device__ __forceinline__ unsigned int encf(float f) {
    unsigned int b = __float_as_uint(f);
    return (b & 0x80000000u) ? ~b : (b | 0x80000000u);
}
__device__ __forceinline__ float decf(unsigned int u) {
    unsigned int b = (u & 0x80000000u) ? (u & 0x7FFFFFFFu) : ~u;
    return __uint_as_float(b);
}

// ---------------------------------------------------------------------------
// WMMA GEMM:  C[M x NT*16] = cvt_bf16(opt_relu(A[M x K])) * B^T (+bias +addsrc)
// A: fp32 row-major, row stride lda (lda % 4 == 0, K % 32 == 0, M % 128 == 0)
// BT: bf16, (NT*16) rows x ldb cols (pre-transposed weights, K zero-padded)
// Workgroup: 256 threads = 8 wave32; tile 128(M) x NT*16(N), K stepped by 32.
// Fragment layouts per CDNA5 ISA 7.12.2:
//   16-bit A 16x32: lanes 0-15 hold K0..7 / K16..23, lanes 16-31 K8..15/K24..31
//   for row M = lane&15; B mirrored N-major; D: lane=N, VGPR index=M (M+8 for
//   lanes 16-31). RELUA is a template flag so the K-loop is branch-free.
// ---------------------------------------------------------------------------
template <int NT, int RELUA>
__global__ __launch_bounds__(256) void gemm_wmma(
    const float* __restrict__ A, int lda,
    const bf16_t* __restrict__ BT, int ldb,
    const float* __restrict__ bias,
    const float* __restrict__ addsrc,
    float* __restrict__ C, int ldc,
    int K, int reluOut)
{
    __shared__ bf16_t ldsA[128 * 32];
    __shared__ bf16_t ldsB[NT * 16 * 32];

    const int tid  = threadIdx.x;
    const int lane = tid & 31;
    const int wave = tid >> 5;            // 0..7
    const int m0   = blockIdx.x * 128;

    v8f zero = {0.f, 0.f, 0.f, 0.f, 0.f, 0.f, 0.f, 0.f};
    v8f acc[NT];
#pragma unroll
    for (int j = 0; j < NT; ++j) acc[j] = zero;

    const int lrow = tid >> 1;            // 0..127
    const int lcol = (tid & 1) * 16;      // 0 or 16

    for (int k0 = 0; k0 < K; k0 += 32) {
        // Stage A tile: 128 rows x 32 K, fp32 -> bf16 (+compile-time relu).
        {
            const float* src = A + (size_t)(m0 + lrow) * lda + k0 + lcol;
            // Prefetch next K-tile of the streamed A operand (global_prefetch_b8).
            __builtin_prefetch(src + 32, 0, 1);
            bf16_t* dst = ldsA + lrow * 32 + lcol;
#pragma unroll
            for (int i = 0; i < 16; i += 4) {
                float4 f = *(const float4*)(src + i);
                if (RELUA) {
                    f.x = fmaxf(f.x, 0.f); f.y = fmaxf(f.y, 0.f);
                    f.z = fmaxf(f.z, 0.f); f.w = fmaxf(f.w, 0.f);
                }
                dst[i + 0] = (bf16_t)f.x; dst[i + 1] = (bf16_t)f.y;
                dst[i + 2] = (bf16_t)f.z; dst[i + 3] = (bf16_t)f.w;
            }
        }
        // Stage B tile: NT*16 rows (columns of B) x 32 K, already bf16.
        if (lrow < NT * 16) {
            const bf16_t* src = BT + (size_t)lrow * ldb + k0 + lcol;
            bf16_t* dst = ldsB + lrow * 32 + lcol;
            *(v8bf*)(dst + 0) = *(const v8bf*)(src + 0);
            *(v8bf*)(dst + 8) = *(const v8bf*)(src + 8);
        }
        __syncthreads();

        const int fr = wave * 16 + (lane & 15);
        const int ko = (lane < 16) ? 0 : 8;
        v8bf alo = *(const v8bf*)(ldsA + fr * 32 + ko);
        v8bf ahi = *(const v8bf*)(ldsA + fr * 32 + ko + 16);
        v16bf afrag = __builtin_shufflevector(alo, ahi,
            0, 1, 2, 3, 4, 5, 6, 7, 8, 9, 10, 11, 12, 13, 14, 15);

#pragma unroll
        for (int j = 0; j < NT; ++j) {
            const int br = j * 16 + (lane & 15);
            v8bf blo = *(const v8bf*)(ldsB + br * 32 + ko);
            v8bf bhi = *(const v8bf*)(ldsB + br * 32 + ko + 16);
            v16bf bfrag = __builtin_shufflevector(blo, bhi,
                0, 1, 2, 3, 4, 5, 6, 7, 8, 9, 10, 11, 12, 13, 14, 15);
            acc[j] = __builtin_amdgcn_wmma_f32_16x16x32_bf16(
                false, afrag, false, bfrag, (short)0, acc[j], false, false);
        }
        __syncthreads();
    }

    // Epilogue (runs once; uniform runtime branches are fine here).
    const int colL = lane & 15;
    const int rowAdd = (lane >> 4) * 8;
#pragma unroll
    for (int j = 0; j < NT; ++j) {
        int n = j * 16 + colL;
        float bv = bias ? bias[n] : 0.0f;
#pragma unroll
        for (int i = 0; i < 8; ++i) {
            int m = m0 + wave * 16 + rowAdd + i;
            float v = acc[j][i] + bv;
            if (addsrc) v += addsrc[(size_t)m * ldc + n];
            if (reluOut) v = fmaxf(v, 0.0f);
            C[(size_t)m * ldc + n] = v;
        }
    }
}

// ---------------------------------------------------------------------------
// Weight prep: fp32 -> bf16, as N x Kpad (zero padded K), two source layouts.
// ---------------------------------------------------------------------------
__global__ void k_transpose_w_bf16(const float* __restrict__ w, bf16_t* __restrict__ o,
                                   int K, int N, int Kpad) {
    int idx = blockIdx.x * 256 + threadIdx.x;
    if (idx >= N * Kpad) return;
    int n = idx / Kpad, k = idx % Kpad;
    o[idx] = (bf16_t)((k < K) ? w[(size_t)k * N + n] : 0.0f);   // w is K x N
}
__global__ void k_convert_w_bf16(const float* __restrict__ w, bf16_t* __restrict__ o,
                                 int K, int N, int Kpad) {
    int idx = blockIdx.x * 256 + threadIdx.x;
    if (idx >= N * Kpad) return;
    int n = idx / Kpad, k = idx % Kpad;
    o[idx] = (bf16_t)((k < K) ? w[(size_t)n * K + k] : 0.0f);   // w is N x K (OIHW)
}

// ---------------------------------------------------------------------------
// Small / bandwidth kernels
// ---------------------------------------------------------------------------
__global__ void k_fill_u32(unsigned int* p, unsigned int v, unsigned int n) {
    unsigned int i = blockIdx.x * 256 + threadIdx.x;
    if (i < n) p[i] = v;
}

// fc_pos: out[m, 0:256] = p[m,0:3] @ w(3x256) + b
__global__ void k_fc_pos(const float* __restrict__ p, const float* __restrict__ w,
                         const float* __restrict__ b, float* __restrict__ out) {
    int idx = blockIdx.x * 256 + threadIdx.x;     // M*64 threads, 4 cols each
    if (idx >= M_ * 64) return;
    int m = idx >> 6, j0 = (idx & 63) * 4;
    float x = p[m * 3 + 0], y = p[m * 3 + 1], z = p[m * 3 + 2];
#pragma unroll
    for (int q = 0; q < 4; ++q) {
        int j = j0 + q;
        out[(size_t)m * 256 + j] = b[j] + x * w[j] + y * w[256 + j] + z * w[512 + j];
    }
}

// per-batch column max over T rows of X (M x 128) -> enc[B*128]
__global__ void k_colmax(const float* __restrict__ X, unsigned int* __restrict__ enc) {
    int b = blockIdx.x, chunk = blockIdx.y, j = threadIdx.x;
    int rows = T_ / gridDim.y;
    size_t base = ((size_t)b * T_ + (size_t)chunk * rows) * 128;
    float mx = -3.402823466e38f;
    for (int r = 0; r < rows; ++r) mx = fmaxf(mx, X[base + (size_t)r * 128 + j]);
    atomicMax(&enc[b * 128 + j], encf(mx));
}

// xcat[m,0:128] = net[m], xcat[m,128:256] = pooled[batch(m)]
__global__ void k_buildcat(const float* __restrict__ net,
                           const unsigned int* __restrict__ pooledEnc,
                           float* __restrict__ xcat) {
    int idx = blockIdx.x * 256 + threadIdx.x;
    if (idx >= M_ * 128) return;
    int m = idx >> 7, j = idx & 127, b = m / T_;
    xcat[(size_t)m * 256 + j]       = net[idx];
    xcat[(size_t)m * 256 + 128 + j] = decf(pooledEnc[b * 128 + j]);
}

// h[b,j] = max_t ( p[b,t] @ w(3x128) + bias )  (encoded atomic max)
__global__ void k_fcp_max(const float* __restrict__ p, const float* __restrict__ w,
                          const float* __restrict__ bias, unsigned int* __restrict__ enc) {
    int b = blockIdx.x, chunk = blockIdx.y, j = threadIdx.x;
    int rows = T_ / gridDim.y;
    const float* pb = p + ((size_t)b * T_ + (size_t)chunk * rows) * 3;
    float mx = -3.402823466e38f;
    for (int r = 0; r < rows; ++r) {
        float v = bias[j] + pb[r * 3] * w[j] + pb[r * 3 + 1] * w[128 + j]
                + pb[r * 3 + 2] * w[256 + j];
        mx = fmaxf(mx, v);
    }
    atomicMax(&enc[b * 128 + j], encf(mx));
}

// Tiny plane MLP (4 rows): 3x relu(128->128), fc_out (128->12),
// and addvec = relu(net_pl) @ fc_plane_hdim (12->128) + b.
__global__ void k_plane_mlp(const unsigned int* __restrict__ hEnc,
                            const float* w0, const float* b0,
                            const float* w1, const float* b1,
                            const float* w2, const float* b2,
                            const float* wo, const float* bo,
                            const float* wph, const float* bph,
                            float* __restrict__ net_pl, float* __restrict__ addvec) {
    __shared__ float sh[128];
    __shared__ float sp[12];
    int j = threadIdx.x;
    const float* Ws[3] = {w0, w1, w2};
    const float* Bs[3] = {b0, b1, b2};
    for (int b = 0; b < 4; ++b) {
        sh[j] = decf(hEnc[b * 128 + j]);
        __syncthreads();
        for (int l = 0; l < 3; ++l) {
            float v = Bs[l][j];
            for (int k = 0; k < 128; ++k) v += sh[k] * Ws[l][k * 128 + j];
            v = fmaxf(v, 0.f);
            __syncthreads();
            sh[j] = v;
            __syncthreads();
        }
        if (j < 12) {
            float v = bo[j];
            for (int k = 0; k < 128; ++k) v += sh[k] * wo[k * 12 + j];
            net_pl[b * 12 + j] = v;
            sp[j] = fmaxf(v, 0.f);
        }
        __syncthreads();
        float av = bph[j];
        for (int k = 0; k < 12; ++k) av += sp[k] * wph[k * 128 + j];
        addvec[b * 128 + j] = av;
        __syncthreads();
    }
}

// _change_basis for the 16 (b,l) planes; writes C_mat (B,NCH,4,3) rows:
// rows 0..2 = inv(R), row 3 = normalizer broadcast.
__global__ void k_change_basis(const float* __restrict__ net_pl, float* __restrict__ Cmat) {
    int i = threadIdx.x;
    if (i >= 16) return;
    int b = i >> 2, l = i & 3;
    const float* nv = net_pl + b * 12 + l * 3;
    float nx = nv[0], ny = nv[1], nz = nv[2];
    float nn = sqrtf(nx * nx + ny * ny + nz * nz);
    nx = nx / nn + 1e-4f; ny = ny / nn + 1e-4f; nz = nz / nn + 1e-4f;
    float v0 = -ny, v1 = nx;                 // v = cross((0,0,1), n), v2 = 0
    float dv = v0 * v0 + v1 * v1;
    float f = (1.0f - nz) / dv;
    float R[3][3];
    R[0][0] = 1.0f - v1 * v1 * f; R[0][1] = v1 * v0 * f;        R[0][2] = v1;
    R[1][0] = v0 * v1 * f;        R[1][1] = 1.0f - v0 * v0 * f; R[1][2] = -v0;
    R[2][0] = -v1;                R[2][1] = v0;                 R[2][2] = 1.0f - dv * f;
    float c00 =  R[1][1] * R[2][2] - R[1][2] * R[2][1];
    float c01 = -(R[1][0] * R[2][2] - R[1][2] * R[2][0]);
    float c02 =  R[1][0] * R[2][1] - R[1][1] * R[2][0];
    float det = R[0][0] * c00 + R[0][1] * c01 + R[0][2] * c02;
    float inv[3][3];
    inv[0][0] = c00 / det;
    inv[0][1] = (R[0][2] * R[2][1] - R[0][1] * R[2][2]) / det;
    inv[0][2] = (R[0][1] * R[1][2] - R[0][2] * R[1][1]) / det;
    inv[1][0] = c01 / det;
    inv[1][1] = (R[0][0] * R[2][2] - R[0][2] * R[2][0]) / det;
    inv[1][2] = (R[0][2] * R[1][0] - R[0][0] * R[1][2]) / det;
    inv[2][0] = c02 / det;
    inv[2][1] = (R[0][1] * R[2][0] - R[0][0] * R[2][1]) / det;
    inv[2][2] = (R[0][0] * R[1][1] - R[0][1] * R[1][0]) / det;
    float bx0 = fabsf(R[0][0]), bx1 = fabsf(R[1][0]), bx2 = fabsf(R[2][0]);
    float by0 = fabsf(R[0][1]), by1 = fabsf(R[1][1]), by2 = fabsf(R[2][1]);
    float sx = bx0 + bx1 + bx2, sxx = bx0 * bx0 + bx1 * bx1 + bx2 * bx2;
    float sy = by0 + by1 + by2, syy = by0 * by0 + by1 * by1 + by2 * by2;
    float cx = (sx / sxx) * sqrtf(sxx);
    float cy = (sy / syy) * sqrtf(syy);
    float nrm = fmaxf(cx, cy);
    float* o = Cmat + (size_t)(b * 4 + l) * 12;
    o[0] = inv[0][0]; o[1] = inv[0][1]; o[2] = inv[0][2];
    o[3] = inv[1][0]; o[4] = inv[1][1]; o[5] = inv[1][2];
    o[6] = inv[2][0]; o[7] = inv[2][1]; o[8] = inv[2][2];
    o[9] = nrm; o[10] = nrm; o[11] = nrm;
}

__global__ void k_add_vec(float* __restrict__ net, const float* __restrict__ addvec) {
    int idx = blockIdx.x * 256 + threadIdx.x;
    if (idx >= M_ * 128) return;
    int m = idx >> 7, j = idx & 127;
    net[idx] += addvec[(m / T_) * 128 + j];
}

// Scatter-max: one thread per (b, l, t); 129 channels per point.
__global__ void k_scatter(const float* __restrict__ p, const float* __restrict__ net,
                          const float* __restrict__ Cmat, unsigned int* __restrict__ planes) {
    int idx = blockIdx.x * 256 + threadIdx.x;
    if (idx >= 4 * 4 * T_) return;
    int t = idx % T_, bl = idx / T_, l = bl & 3, b = bl >> 2;
    size_t m = (size_t)b * T_ + t;
    const float inv_md = 1.0f / 0.55f;
    float px = p[m * 3 + 0] * inv_md, py = p[m * 3 + 1] * inv_md, pz = p[m * 3 + 2] * inv_md;
    const float* Cm = Cmat + (size_t)(b * 4 + l) * 12;
    float nrm = Cm[9] + 0.05f;
    float pr0 = (Cm[0] * px + Cm[1] * py + Cm[2] * pz) / nrm;
    float pr1 = (Cm[3] * px + Cm[4] * py + Cm[5] * pz) / nrm;
    float pr2 = (Cm[6] * px + Cm[7] * py + Cm[8] * pz) / nrm;
    const float INV_INT = 63.0f / 2.0f;           // 1 / (2/(GRID-1))
    float x = (pr0 + 1.0f) * INV_INT;
    float y = (pr1 + 1.0f) * INV_INT;
    x = (x >= 63.0f) ? 62.9f : x; x = (x < 0.0f) ? 0.0f : x;
    y = (y >= 63.0f) ? 62.9f : y; y = (y < 0.0f) ? 0.0f : y;
    int xi = (int)rintf(x), yi = (int)rintf(y);
    int cell = xi + 64 * yi;
    unsigned int* dst = planes + (((size_t)(b * 4 + l) * 4096 + cell) * 129);
    const float* nf = net + m * 128;
    for (int c = 0; c < 128; ++c) atomicMax(&dst[c], encf(nf[c]));
    atomicMax(&dst[128], encf(pr2));
}

// Decode planes into UNet input X0, HWC per image, n = b*5 + l (l==4 -> zeros).
__global__ void k_build_x0(const unsigned int* __restrict__ planes, float* __restrict__ X0) {
    size_t idx = (size_t)blockIdx.x * 256 + threadIdx.x;
    size_t total = (size_t)20 * 4096 * 129;
    if (idx >= total) return;
    int c = (int)(idx % 129);
    size_t r = idx / 129;
    int pix = (int)(r % 4096);
    int n = (int)(r / 4096);
    int b = n / 5, l = n % 5;
    X0[idx] = (l < 4) ? decf(planes[((size_t)(b * 4 + l) * 4096 + pix) * 129 + c]) : 0.0f;
}

// im2col for one HWC image: col[pix, k] with k = ci*9 + (ky*3+kx), SAME pad.
__global__ void k_im2col(const float* __restrict__ X, float* __restrict__ col,
                         int H, int W, int Cin, int Kpad) {
    size_t idx = (size_t)blockIdx.x * 256 + threadIdx.x;
    size_t total = (size_t)H * W * Kpad;
    if (idx >= total) return;
    int k = (int)(idx % Kpad);
    int pix = (int)(idx / Kpad);
    float v = 0.0f;
    int K = Cin * 9;
    if (k < K) {
        int ci = k / 9, tap = k % 9;
        int y = pix / W + tap / 3 - 1;
        int x = pix % W + tap % 3 - 1;
        if (y >= 0 && y < H && x >= 0 && x < W)
            v = X[((size_t)y * W + x) * Cin + ci];
    }
    col[idx] = v;
}

__global__ void k_pool2(const float* __restrict__ X, float* __restrict__ Y,
                        int H, int W, int C, int Nimg) {
    int Ho = H / 2, Wo = W / 2;
    size_t idx = (size_t)blockIdx.x * 256 + threadIdx.x;
    size_t total = (size_t)Nimg * Ho * Wo * C;
    if (idx >= total) return;
    int c = (int)(idx % C);
    size_t r = idx / C;
    int opix = (int)(r % (Ho * Wo));
    int n = (int)(r / (Ho * Wo));
    int oy = opix / Wo, ox = opix % Wo;
    const float* Xi = X + (size_t)n * H * W * C;
    int y = 2 * oy, x = 2 * ox;
    float a = Xi[((size_t)y * W + x) * C + c];
    float b = Xi[((size_t)y * W + x + 1) * C + c];
    float d = Xi[((size_t)(y + 1) * W + x) * C + c];
    float e = Xi[((size_t)(y + 1) * W + x + 1) * C + c];
    Y[idx] = fmaxf(fmaxf(a, b), fmaxf(d, e));
}

__global__ void k_up2(const float* __restrict__ X, float* __restrict__ Y,
                      int H, int W, int C, int Nimg) {
    size_t idx = (size_t)blockIdx.x * 256 + threadIdx.x;
    size_t total = (size_t)Nimg * H * W * 4 * C;
    if (idx >= total) return;
    int c = (int)(idx % C);
    size_t r = idx / C;
    int opix = (int)(r % (H * W * 4));
    int n = (int)(r / (H * W * 4));
    int W2 = 2 * W;
    int oy = opix / W2, ox = opix % W2;
    Y[idx] = X[((size_t)n * H * W + (size_t)(oy >> 1) * W + (ox >> 1)) * C + c];
}

__global__ void k_copy_ch(const float* __restrict__ src, float* __restrict__ dst,
                          int HW, int Cs, int Cd, int off, int Nimg) {
    size_t idx = (size_t)blockIdx.x * 256 + threadIdx.x;
    size_t total = (size_t)Nimg * HW * Cs;
    if (idx >= total) return;
    int c = (int)(idx % Cs);
    size_t r = idx / Cs;
    int pix = (int)(r % HW);
    int n = (int)(r / HW);
    dst[((size_t)n * HW + pix) * Cd + off + c] = src[idx];
}

// ---------------------------------------------------------------------------
// Host helpers
// ---------------------------------------------------------------------------
static inline void launch_gemm(hipStream_t s, const float* A, int lda,
                               const bf16_t* BT, int ldb,
                               const float* bias, const float* addsrc,
                               float* C, int ldc, int M, int N, int K,
                               int reluA, int reluOut) {
    dim3 grid(M / 128);
    if (N == 32) {
        if (reluA) gemm_wmma<2, 1><<<grid, 256, 0, s>>>(A, lda, BT, ldb, bias, addsrc, C, ldc, K, reluOut);
        else       gemm_wmma<2, 0><<<grid, 256, 0, s>>>(A, lda, BT, ldb, bias, addsrc, C, ldc, K, reluOut);
    } else if (N == 64) {
        if (reluA) gemm_wmma<4, 1><<<grid, 256, 0, s>>>(A, lda, BT, ldb, bias, addsrc, C, ldc, K, reluOut);
        else       gemm_wmma<4, 0><<<grid, 256, 0, s>>>(A, lda, BT, ldb, bias, addsrc, C, ldc, K, reluOut);
    } else {
        if (reluA) gemm_wmma<8, 1><<<grid, 256, 0, s>>>(A, lda, BT, ldb, bias, addsrc, C, ldc, K, reluOut);
        else       gemm_wmma<8, 0><<<grid, 256, 0, s>>>(A, lda, BT, ldb, bias, addsrc, C, ldc, K, reluOut);
    }
}

static inline void conv3x3(hipStream_t s, const float* X, int Nimg, int H, int W,
                           int Cin, const bf16_t* Wbt, const float* bias,
                           float* Y, int Cout, float* colbuf) {
    int K = Cin * 9, Kpad = (K + 31) & ~31, HW = H * W;
    size_t tot = (size_t)HW * Kpad;
    unsigned int gi = (unsigned int)((tot + 255) / 256);
    for (int n = 0; n < Nimg; ++n) {
        k_im2col<<<gi, 256, 0, s>>>(X + (size_t)n * HW * Cin, colbuf, H, W, Cin, Kpad);
        launch_gemm(s, colbuf, Kpad, Wbt, Kpad, bias, nullptr,
                    Y + (size_t)n * HW * Cout, Cout, HW, Cout, Kpad, 0, /*relu*/1);
    }
}

// ---------------------------------------------------------------------------
// Driver. Input order assumes JAX pytree flatten (dict keys sorted) of
// {'p', 'params'} with nested {'b','w'} per linear/conv.
// ---------------------------------------------------------------------------
extern "C" void kernel_launch(void* const* d_in, const int* in_sizes, int n_in,
                              void* d_out, int out_size, void* d_ws, size_t ws_size,
                              hipStream_t stream) {
    (void)in_sizes; (void)n_in; (void)out_size; (void)ws_size;
    const float* p = (const float*)d_in[0];
    struct Blk { const float *b0, *w0, *b1, *w1, *sc; };
    Blk blk[5];
    for (int i = 0; i < 5; ++i) {
        blk[i].b0 = (const float*)d_in[1 + i * 5];
        blk[i].w0 = (const float*)d_in[2 + i * 5];
        blk[i].b1 = (const float*)d_in[3 + i * 5];
        blk[i].w1 = (const float*)d_in[4 + i * 5];
        blk[i].sc = (const float*)d_in[5 + i * 5];
    }
    const float* fph_b = (const float*)d_in[26];
    const float* fph_w = (const float*)d_in[27];
    const float* fpos_b = (const float*)d_in[28];
    const float* fpos_w = (const float*)d_in[29];
    const float* pn0_b = (const float*)d_in[30]; const float* pn0_w = (const float*)d_in[31];
    const float* pn1_b = (const float*)d_in[32]; const float* pn1_w = (const float*)d_in[33];
    const float* pn2_b = (const float*)d_in[34]; const float* pn2_w = (const float*)d_in[35];
    const float* pno_b = (const float*)d_in[36]; const float* pno_w = (const float*)d_in[37];
    const float* pnp_b = (const float*)d_in[38]; const float* pnp_w = (const float*)d_in[39];
    const float* d1a_b = (const float*)d_in[40]; const float* d1a_w = (const float*)d_in[41];
    const float* d1b_b = (const float*)d_in[42]; const float* d1b_w = (const float*)d_in[43];
    const float* d2a_b = (const float*)d_in[44]; const float* d2a_w = (const float*)d_in[45];
    const float* d2b_b = (const float*)d_in[46]; const float* d2b_w = (const float*)d_in[47];
    const float* e1a_b = (const float*)d_in[48]; const float* e1a_w = (const float*)d_in[49];
    const float* e1b_b = (const float*)d_in[50]; const float* e1b_w = (const float*)d_in[51];
    const float* e2a_b = (const float*)d_in[52]; const float* e2a_w = (const float*)d_in[53];
    const float* e2b_b = (const float*)d_in[54]; const float* e2b_w = (const float*)d_in[55];
    const float* e3a_b = (const float*)d_in[56]; const float* e3a_w = (const float*)d_in[57];
    const float* e3b_b = (const float*)d_in[58]; const float* e3b_w = (const float*)d_in[59];
    const float* fin_b = (const float*)d_in[60]; const float* fin_w = (const float*)d_in[61];
    const float* u1_b  = (const float*)d_in[62]; const float* u1_w  = (const float*)d_in[63];
    const float* u2_b  = (const float*)d_in[64]; const float* u2_w  = (const float*)d_in[65];

    char* wsb = (char*)d_ws;
    size_t cur = 0;
    auto alloc = [&](size_t bytes) -> void* {
        void* q = (void*)(wsb + cur);
        cur = (cur + bytes + 255) & ~(size_t)255;
        return q;
    };
    auto prep_fc = [&](const float* w, int K, int N) -> bf16_t* {
        bf16_t* o = (bf16_t*)alloc((size_t)N * K * 2);
        int tot = N * K;
        k_transpose_w_bf16<<<(tot + 255) / 256, 256, 0, stream>>>(w, o, K, N, K);
        return o;
    };
    auto prep_conv = [&](const float* w, int K, int N) -> bf16_t* {
        int Kpad = (K + 31) & ~31;
        bf16_t* o = (bf16_t*)alloc((size_t)N * Kpad * 2);
        int tot = N * Kpad;
        k_convert_w_bf16<<<(tot + 255) / 256, 256, 0, stream>>>(w, o, K, N, Kpad);
        return o;
    };

    // --- persistent region: bf16 weights + long-lived activations ----------
    bf16_t *bw0[5], *bw1[5], *bsc[5];
    for (int i = 0; i < 5; ++i) {
        bw0[i] = prep_fc(blk[i].w0, 256, 128);
        bw1[i] = prep_fc(blk[i].w1, 128, 128);
        bsc[i] = prep_fc(blk[i].sc, 256, 128);
    }
    bf16_t* bt_e1a = prep_conv(e1a_w, 129 * 9, 32);
    bf16_t* bt_e1b = prep_conv(e1b_w, 32 * 9, 32);
    bf16_t* bt_e2a = prep_conv(e2a_w, 32 * 9, 64);
    bf16_t* bt_e2b = prep_conv(e2b_w, 64 * 9, 64);
    bf16_t* bt_e3a = prep_conv(e3a_w, 64 * 9, 128);
    bf16_t* bt_e3b = prep_conv(e3b_w, 128 * 9, 128);
    bf16_t* bt_u2  = prep_conv(u2_w, 128, 64);
    bf16_t* bt_d2a = prep_conv(d2a_w, 128 * 9, 64);
    bf16_t* bt_d2b = prep_conv(d2b_w, 64 * 9, 64);
    bf16_t* bt_u1  = prep_conv(u1_w, 64, 32);
    bf16_t* bt_d1a = prep_conv(d1a_w, 64 * 9, 32);
    bf16_t* bt_d1b = prep_conv(d1b_w, 32 * 9, 32);
    bf16_t* bt_fin = prep_conv(fin_w, 32, 128);

    float* netF = (float*)alloc((size_t)M_ * 128 * 4);
    unsigned int* planesEnc = (unsigned int*)alloc((size_t)16 * 4096 * 129 * 4);
    unsigned int* pooledEnc = (unsigned int*)alloc(512 * 4);
    unsigned int* hEnc = (unsigned int*)alloc(512 * 4);
    float* net_pl = (float*)alloc(48 * 4);
    float* addvec = (float*)alloc(512 * 4);
    size_t mark = cur;

    // --- point-MLP phase scratch -------------------------------------------
    float* xcat = (float*)alloc((size_t)M_ * 256 * 4);
    float* Hf   = (float*)alloc((size_t)M_ * 128 * 4);
    float* Df   = (float*)alloc((size_t)M_ * 128 * 4);

    k_fc_pos<<<M_ * 64 / 256, 256, 0, stream>>>(p, fpos_w, fpos_b, xcat);
    for (int b = 0; b < 5; ++b) {
        // h  = relu(xcat) @ W0 + b0
        launch_gemm(stream, xcat, 256, bw0[b], 256, blk[b].b0, nullptr, Hf, 128, M_, 128, 256, 1, 0);
        // dx = relu(h) @ W1 + b1
        launch_gemm(stream, Hf, 128, bw1[b], 128, blk[b].b1, nullptr, Df, 128, M_, 128, 128, 1, 0);
        // net = xcat @ SC + dx
        launch_gemm(stream, xcat, 256, bsc[b], 256, nullptr, Df, netF, 128, M_, 128, 256, 0, 0);
        if (b < 4) {
            k_fill_u32<<<2, 256, 0, stream>>>(pooledEnc, 0u, 512);
            k_colmax<<<dim3(4, 64), 128, 0, stream>>>(netF, pooledEnc);
            k_buildcat<<<M_ * 128 / 256, 256, 0, stream>>>(netF, pooledEnc, xcat);
        }
    }

    // --- plane branch -------------------------------------------------------
    k_fill_u32<<<2, 256, 0, stream>>>(hEnc, 0u, 512);
    k_fcp_max<<<dim3(4, 64), 128, 0, stream>>>(p, pnp_w, pnp_b, hEnc);
    k_plane_mlp<<<1, 128, 0, stream>>>(hEnc, pn0_w, pn0_b, pn1_w, pn1_b, pn2_w, pn2_b,
                                       pno_w, pno_b, fph_w, fph_b, net_pl, addvec);
    float* CmatOut = (float*)d_out + COUT_ELEMS;
    k_change_basis<<<1, 32, 0, stream>>>(net_pl, CmatOut);
    k_add_vec<<<M_ * 128 / 256, 256, 0, stream>>>(netF, addvec);

    // --- scatter-max onto planes -------------------------------------------
    unsigned int planeWords = 16u * 4096u * 129u;
    k_fill_u32<<<(planeWords + 255) / 256, 256, 0, stream>>>(planesEnc, 0x80000000u, planeWords);
    k_scatter<<<4 * 4 * T_ / 256, 256, 0, stream>>>(p, netF, CmatOut, planesEnc);

    // --- UNet phase (reuse point scratch region) ---------------------------
    cur = mark;
    float* X0    = (float*)alloc((size_t)20 * 4096 * 129 * 4);
    float* t1    = (float*)alloc((size_t)20 * 4096 * 32 * 4);
    float* e1    = (float*)alloc((size_t)20 * 4096 * 32 * 4);
    float* p1    = (float*)alloc((size_t)20 * 1024 * 32 * 4);
    float* t2    = (float*)alloc((size_t)20 * 1024 * 64 * 4);
    float* e2    = (float*)alloc((size_t)20 * 1024 * 64 * 4);
    float* p2    = (float*)alloc((size_t)20 * 256 * 64 * 4);
    float* t3    = (float*)alloc((size_t)20 * 256 * 128 * 4);
    float* btA   = (float*)alloc((size_t)20 * 256 * 128 * 4);
    float* ub2   = (float*)alloc((size_t)20 * 1024 * 128 * 4);
    float* d2cat = (float*)alloc((size_t)20 * 1024 * 128 * 4);
    float* t4    = (float*)alloc((size_t)20 * 1024 * 64 * 4);
    float* d2    = (float*)alloc((size_t)20 * 1024 * 64 * 4);
    float* ub1   = (float*)alloc((size_t)20 * 4096 * 64 * 4);
    float* d1cat = (float*)alloc((size_t)20 * 4096 * 64 * 4);
    float* t5    = (float*)alloc((size_t)20 * 4096 * 32 * 4);
    float* d1    = (float*)alloc((size_t)20 * 4096 * 32 * 4);
    float* colbuf = (float*)alloc((size_t)4096 * 1184 * 4);

    {
        size_t tot = (size_t)20 * 4096 * 129;
        k_build_x0<<<(unsigned int)((tot + 255) / 256), 256, 0, stream>>>(planesEnc, X0);
    }
    conv3x3(stream, X0, 20, 64, 64, 129, bt_e1a, e1a_b, t1, 32, colbuf);
    conv3x3(stream, t1, 20, 64, 64, 32, bt_e1b, e1b_b, e1, 32, colbuf);
    k_pool2<<<(unsigned int)(((size_t)20 * 1024 * 32 + 255) / 256), 256, 0, stream>>>(e1, p1, 64, 64, 32, 20);
    conv3x3(stream, p1, 20, 32, 32, 32, bt_e2a, e2a_b, t2, 64, colbuf);
    conv3x3(stream, t2, 20, 32, 32, 64, bt_e2b, e2b_b, e2, 64, colbuf);
    k_pool2<<<(unsigned int)(((size_t)20 * 256 * 64 + 255) / 256), 256, 0, stream>>>(e2, p2, 32, 32, 64, 20);
    conv3x3(stream, p2, 20, 16, 16, 64, bt_e3a, e3a_b, t3, 128, colbuf);
    conv3x3(stream, t3, 20, 16, 16, 128, bt_e3b, e3b_b, btA, 128, colbuf);
    k_up2<<<(unsigned int)(((size_t)20 * 1024 * 128 + 255) / 256), 256, 0, stream>>>(btA, ub2, 16, 16, 128, 20);
    // u2 1x1 conv (relu) into d2cat cols 0..63; e2 copied into cols 64..127
    launch_gemm(stream, ub2, 128, bt_u2, 128, u2_b, nullptr, d2cat, 128, 20 * 1024, 64, 128, 0, 1);
    k_copy_ch<<<(unsigned int)(((size_t)20 * 1024 * 64 + 255) / 256), 256, 0, stream>>>(e2, d2cat, 1024, 64, 128, 64, 20);
    conv3x3(stream, d2cat, 20, 32, 32, 128, bt_d2a, d2a_b, t4, 64, colbuf);
    conv3x3(stream, t4, 20, 32, 32, 64, bt_d2b, d2b_b, d2, 64, colbuf);
    k_up2<<<(unsigned int)(((size_t)20 * 4096 * 64 + 255) / 256), 256, 0, stream>>>(d2, ub1, 32, 32, 64, 20);
    launch_gemm(stream, ub1, 64, bt_u1, 64, u1_b, nullptr, d1cat, 64, 20 * 4096, 32, 64, 0, 1);
    k_copy_ch<<<(unsigned int)(((size_t)20 * 4096 * 32 + 255) / 256), 256, 0, stream>>>(e1, d1cat, 4096, 32, 64, 32, 20);
    conv3x3(stream, d1cat, 20, 64, 64, 64, bt_d1a, d1a_b, t5, 32, colbuf);
    conv3x3(stream, t5, 20, 64, 64, 32, bt_d1b, d1b_b, d1, 32, colbuf);
    // final 1x1 conv -> d_out directly (HWC == required (B,L,64,64,C) layout)
    launch_gemm(stream, d1, 32, bt_fin, 32, fin_b, nullptr, (float*)d_out, 128, 20 * 4096, 128, 32, 0, 0);
}
